// TemporalAttention_6141803233367
// MI455X (gfx1250) — matrix operand order repaired and checked by reading, not compiled, not run on hardware
//
#include <hip/hip_runtime.h>
#include <hip/hip_bf16.h>

// ---------------------------------------------------------------------------
// TemporalAttention for MI455X (gfx1250, wave32, WMMA bf16)
//   x:(B=2,T=128,S=256,D=1024)  H=16 HKV=4 HD=64  -> out (B,T,S,D) f32
// Pipeline: prep(weights->bf16) -> qkv(GEMM+RMSnorm+RoPE) -> attn -> outproj
// Fragment loads use 16B-aligned LDS rows so each WMMA fragment is
// exactly 2x ds_load_b128.
// ---------------------------------------------------------------------------

typedef __attribute__((ext_vector_type(16))) __bf16 v16bf;
typedef __attribute__((ext_vector_type(8)))  float  v8f;
typedef unsigned short u16;
typedef unsigned int   u32;

constexpr int C_DIM = 1024;
constexpr int C_H   = 16;
constexpr int C_HKV = 4;
constexpr int C_HD  = 64;
constexpr int C_T   = 128;
constexpr int C_S   = 256;
constexpr int C_N   = 512;          // B*S
constexpr float C_SCALE = 0.125f;   // 1/sqrt(64)
constexpr float C_CAP   = 50.0f;
constexpr float C_EPS   = 1e-6f;
constexpr float C_LN_ROPE = 9.210340371976184f; // ln(10000)

// workspace layout (bytes)
constexpr size_t OFF_WQKV = 0;                            // 1536*1024 bf16
constexpr size_t OFF_WO   = OFF_WQKV + 1536ull*1024*2;    // 1024*1024 bf16
constexpr size_t OFF_Q    = OFF_WO   + 1024ull*1024*2;    // N*H*T*HD bf16
constexpr size_t OFF_K    = OFF_Q    + 512ull*16*128*64*2;
constexpr size_t OFF_VT   = OFF_K    + 512ull*4*128*64*2; // V transposed [n][hkv][HD][T]
constexpr size_t OFF_AO   = OFF_VT   + 512ull*4*128*64*2; // attn out [n][t][1024] bf16

__device__ __forceinline__ u16 f2bf(float x) {
    u32 u = __float_as_uint(x);
    u32 r = 0x7FFFu + ((u >> 16) & 1u);
    return (u16)((u + r) >> 16);
}
__device__ __forceinline__ u32 pack2bf(float lo, float hi) {
    return (u32)f2bf(lo) | ((u32)f2bf(hi) << 16);
}

// A-fragment 16x32 bf16 from row-major tile [M][K] (ISA 16-bit A layout).
// lane r=lane&15 holds row M=r; h=lane>>4.
// VGPRs 0..3 hold K = h*8 .. h*8+7 (16 contiguous bytes),
// VGPRs 4..7 hold K = 16+h*8 .. 16+h*8+7.
// Requires lda % 8 == 0 (16B-aligned rows) -> two ds_load_b128.
__device__ __forceinline__ v16bf ld_frag_a(const u16* tile, int lda, int lane) {
    int r = lane & 15, h = lane >> 4;
    const u16* row = tile + r * lda + h * 8;
    union { v16bf v; uint4 q[2]; } f;
    f.q[0] = *(const uint4*)(row);
    f.q[1] = *(const uint4*)(row + 16);
    return f.v;
}

// B-fragment 32x16 bf16 from N-major tile [N][K] (ISA B layout).
// lane n=lane&15 holds column N=n; kb=(lane>>4)*8; VGPRs 0..3: K=kb..kb+7,
// VGPRs 4..7: K=16+kb..16+kb+7 -> two ds_load_b128.
__device__ __forceinline__ v16bf ld_frag_b(const u16* tile, int lda, int lane) {
    int nidx = lane & 15, kb = (lane >> 4) * 8;
    const u16* row = tile + nidx * lda + kb;
    union { v16bf v; uint4 q[2]; } f;
    f.q[0] = *(const uint4*)(row);
    f.q[1] = *(const uint4*)(row + 16);
    return f.v;
}

__device__ __forceinline__ v8f wmma_bf16(v16bf a, v16bf b, v8f c) {
    return __builtin_amdgcn_wmma_f32_16x16x32_bf16(false, a, false, b, (short)0, c, false, false);
}

__device__ __forceinline__ float sum16(float x) {
    x += __shfl_xor(x, 1, 32); x += __shfl_xor(x, 2, 32);
    x += __shfl_xor(x, 4, 32); x += __shfl_xor(x, 8, 32);
    return x;
}
__device__ __forceinline__ float max16(float x) {
    x = fmaxf(x, __shfl_xor(x, 1, 32)); x = fmaxf(x, __shfl_xor(x, 2, 32));
    x = fmaxf(x, __shfl_xor(x, 4, 32)); x = fmaxf(x, __shfl_xor(x, 8, 32));
    return x;
}

// ---------------------------------------------------------------------------
// Kernel 1: weights f32 -> bf16 (Wq|Wk|Wv fused rows 0..1535, Wo separate)
// ---------------------------------------------------------------------------
__global__ void prep_kernel(const float* __restrict__ Wq, const float* __restrict__ Wk,
                            const float* __restrict__ Wv, const float* __restrict__ Wo,
                            u16* __restrict__ wqkv, u16* __restrict__ wo) {
    size_t i = (size_t)blockIdx.x * 256 + threadIdx.x;
    const size_t NQ = 1048576, NK = 262144, NV = 262144;
    if (i < NQ)                     wqkv[i] = f2bf(Wq[i]);
    else if (i < NQ + NK)           wqkv[i] = f2bf(Wk[i - NQ]);
    else if (i < NQ + NK + NV)      wqkv[i] = f2bf(Wv[i - NQ - NK]);
    else if (i < NQ + NK + NV + NQ) wo[i - NQ - NK - NV] = f2bf(Wo[i - NQ - NK - NV]);
}

// ---------------------------------------------------------------------------
// Kernel 2: QKV projection + QK RMS-norm + RoPE.
// grid (n=512, cb=6): cb 0..3 -> Q cols, 4 -> K, 5 -> V.  WG: 256 thr = 8 waves.
// Each WG: 128x1024 @ 1024x256 -> 128x256 f32, WMMA bf16, then epilogue.
// ---------------------------------------------------------------------------
__global__ __launch_bounds__(256) void qkv_kernel(
        const float* __restrict__ x, const u16* __restrict__ wqkv,
        const float* __restrict__ qg, const float* __restrict__ kg,
        u16* __restrict__ qbuf, u16* __restrict__ kbuf, u16* __restrict__ vtbuf) {
    const int n  = blockIdx.x;
    const int cb = blockIdx.y;
    const int tid = threadIdx.x, lane = tid & 31, w = tid >> 5;
    const int half = lane >> 4, r = lane & 15;
    const int b = n >> 8, s = n & 255;

    const int LDX = 40, LDW = 40;        // 16B-aligned rows, conflict-free banks
    __shared__ u16 Xt[128 * 40];
    __shared__ u16 Wt[256 * 40];

    const float* xb = x + ((size_t)(b * C_T) * C_S + s) * C_DIM; // + t*S*D + k

    v8f acc[16];
#pragma unroll
    for (int nt = 0; nt < 16; ++nt) acc[nt] = (v8f){0,0,0,0,0,0,0,0};

    const int colbase = cb * 256;
    const int xrow = tid >> 1, xc0 = (tid & 1) * 16;
    const float* xsrc_base = xb + (size_t)xrow * (C_S * C_DIM) + xc0;
    const u16*   wsrc_base = wqkv + (size_t)(colbase + tid) * C_DIM;

    for (int k0 = 0; k0 < C_DIM; k0 += 32) {
        { // X tile: 128 rows x 32 f32 -> bf16 LDS (packed b128 stores)
            const float* src = xsrc_base + k0;
            u32 p[8];
#pragma unroll
            for (int i = 0; i < 16; i += 4) {
                float4 f = *(const float4*)(src + i);
                p[i/2]     = pack2bf(f.x, f.y);
                p[i/2 + 1] = pack2bf(f.z, f.w);
            }
            uint4* dst = (uint4*)&Xt[xrow * LDX + xc0];
            dst[0] = make_uint4(p[0], p[1], p[2], p[3]);
            dst[1] = make_uint4(p[4], p[5], p[6], p[7]);
        }
        { // W tile: 256 rows x 32 bf16 (N-major), b128 copies
            const uint4* src = (const uint4*)(wsrc_base + k0);
            uint4* dst = (uint4*)&Wt[tid * LDW];
#pragma unroll
            for (int i = 0; i < 4; ++i) dst[i] = src[i];
        }
        if (k0 + 32 < C_DIM) { // prefetch next tiles (global_prefetch_b8)
            __builtin_prefetch((const void*)(xsrc_base + k0 + 32), 0, 1);
            __builtin_prefetch((const void*)(wsrc_base + k0 + 32), 0, 1);
        }
        __syncthreads();
        v16bf a = ld_frag_a(&Xt[(w * 16) * LDX], LDX, lane);
#pragma unroll
        for (int nt = 0; nt < 16; ++nt) {
            v16bf bf = ld_frag_b(&Wt[(nt * 16) * LDW], LDW, lane);
            acc[nt] = wmma_bf16(a, bf, acc[nt]);
        }
        __syncthreads();
    }

    // ---- epilogue ----
    if (cb <= 4) {
        // Q (cb<4) or K (cb==4): RMS-norm over each 64-wide head, then RoPE
        const float* gamma = (cb == 4) ? kg : qg;
#pragma unroll
        for (int hh = 0; hh < 4; ++hh) {
#pragma unroll
            for (int v = 0; v < 8; ++v) {
                float ss = 0.f;
#pragma unroll
                for (int j = 0; j < 4; ++j) { float t = acc[hh*4 + j][v]; ss += t * t; }
                ss = sum16(ss);
                float rms = rsqrtf(ss * (1.0f / 64.0f) + C_EPS);
                int t_idx = w * 16 + half * 8 + v;
                float nv[4];
#pragma unroll
                for (int j = 0; j < 4; ++j) {
                    int d = j * 16 + r;
                    nv[j] = acc[hh*4 + j][v] * rms * gamma[d];
                }
#pragma unroll
                for (int j = 0; j < 4; ++j) {
                    int d = j * 16 + r;
                    int dm = d & 31;
                    float ang = (float)t_idx * __expf(-(float)dm * (C_LN_ROPE / 32.0f));
                    float c = cosf(ang), sn = sinf(ang);
                    float rot = (j < 2) ? -nv[j + 2] : nv[j - 2];
                    u16 o = f2bf(nv[j] * c + rot * sn);
                    if (cb < 4) {
                        int hq = cb * 4 + hh;
                        qbuf[(((size_t)n * C_H + hq) * C_T + t_idx) * C_HD + d] = o;
                    } else {
                        kbuf[(((size_t)n * C_HKV + hh) * C_T + t_idx) * C_HD + d] = o;
                    }
                }
            }
        }
    } else {
        // V: store transposed [n][hkv][HD][T]
#pragma unroll
        for (int nt = 0; nt < 16; ++nt) {
            int col = nt * 16 + r, h2 = col >> 6, hd = col & 63;
#pragma unroll
            for (int v = 0; v < 8; ++v) {
                int t_idx = w * 16 + half * 8 + v;
                vtbuf[(((size_t)n * C_HKV + h2) * C_HD + hd) * C_T + t_idx] = f2bf(acc[nt][v]);
            }
        }
    }
}

// ---------------------------------------------------------------------------
// Kernel 3: attention per (n, head). S=Q·K^T (WMMA), softcap+causal softmax,
// O=P·V (WMMA). 256 thr = 8 waves, wave w owns query rows [16w,16w+16).
// ---------------------------------------------------------------------------
__global__ __launch_bounds__(256) void attn_kernel(
        const u16* __restrict__ qbuf, const u16* __restrict__ kbuf,
        const u16* __restrict__ vtbuf, u16* __restrict__ attnout) {
    const int n = blockIdx.x, h = blockIdx.y, hkv = h >> 2;
    const int tid = threadIdx.x, lane = tid & 31, w = tid >> 5;
    const int half = lane >> 4, r = lane & 15;

    const int LQ = 72, LV = 136;         // 16B-aligned rows
    __shared__ u16 Qs[128 * 72];
    __shared__ u16 Ks[128 * 72];
    __shared__ u16 Vs[64 * 136];
    __shared__ u16 Ps[128 * 136];        // probabilities, bf16

    const u16* qsrc = qbuf  + (((size_t)n * C_H   + h  ) * C_T) * C_HD;
    const u16* ksrc = kbuf  + (((size_t)n * C_HKV + hkv) * C_T) * C_HD;
    const u16* vsrc = vtbuf + (((size_t)n * C_HKV + hkv) * C_HD) * C_T;

    { // loads: Q/K 128x64, Vt 64x128 (b128 copies)
        int row = tid >> 1, c0 = (tid & 1) * 32;
        const uint4* qp = (const uint4*)(qsrc + row * C_HD + c0);
        const uint4* kp = (const uint4*)(ksrc + row * C_HD + c0);
        uint4* qd = (uint4*)&Qs[row * LQ + c0];
        uint4* kd = (uint4*)&Ks[row * LQ + c0];
#pragma unroll
        for (int i = 0; i < 4; ++i) { qd[i] = qp[i]; kd[i] = kp[i]; }
        int vr = tid >> 2, vc = (tid & 3) * 32;
        const uint4* vp = (const uint4*)(vsrc + vr * C_T + vc);
        uint4* vd = (uint4*)&Vs[vr * LV + vc];
#pragma unroll
        for (int i = 0; i < 4; ++i) vd[i] = vp[i];
    }
    __syncthreads();

    // S = Q @ K^T  (16x128 per wave)
    v8f sacc[8];
#pragma unroll
    for (int kt = 0; kt < 8; ++kt) sacc[kt] = (v8f){0,0,0,0,0,0,0,0};
#pragma unroll
    for (int ks = 0; ks < 2; ++ks) {
        v16bf a = ld_frag_a(&Qs[(w * 16) * LQ + ks * 32], LQ, lane);
#pragma unroll
        for (int kt = 0; kt < 8; ++kt) {
            v16bf bf = ld_frag_b(&Ks[(kt * 16) * LQ + ks * 32], LQ, lane);
            sacc[kt] = wmma_bf16(a, bf, sacc[kt]);
        }
    }

    // softcap + causal mask + softmax per query row; write P (bf16) to LDS
#pragma unroll
    for (int v = 0; v < 8; ++v) {
        int qi = w * 16 + half * 8 + v;
        float mx = -1e30f;
#pragma unroll
        for (int kt = 0; kt < 8; ++kt) {
            float sv = sacc[kt][v] * C_SCALE;
            sv = C_CAP * tanhf(sv * (1.0f / C_CAP));
            int key = kt * 16 + r;
            if (key > qi) sv = -1e30f;
            sacc[kt][v] = sv;
            mx = fmaxf(mx, sv);
        }
        mx = max16(mx);
        float sum = 0.f;
#pragma unroll
        for (int kt = 0; kt < 8; ++kt) {
            float e = __expf(sacc[kt][v] - mx);
            sacc[kt][v] = e;
            sum += e;
        }
        sum = sum16(sum);
        float inv = 1.0f / sum;
        u16* prow = &Ps[(size_t)qi * LV];
#pragma unroll
        for (int kt = 0; kt < 8; ++kt)
            prow[kt * 16 + r] = f2bf(sacc[kt][v] * inv);
    }
    __syncthreads();

    // O = P @ V  (16x64 per wave)
    v8f oacc[4];
#pragma unroll
    for (int ht = 0; ht < 4; ++ht) oacc[ht] = (v8f){0,0,0,0,0,0,0,0};
#pragma unroll
    for (int ks = 0; ks < 4; ++ks) {
        v16bf a = ld_frag_a(&Ps[(w * 16) * LV + ks * 32], LV, lane);
#pragma unroll
        for (int ht = 0; ht < 4; ++ht) {
            v16bf bf = ld_frag_b(&Vs[(ht * 16) * LV + ks * 32], LV, lane);
            oacc[ht] = wmma_bf16(a, bf, oacc[ht]);
        }
    }
#pragma unroll
    for (int ht = 0; ht < 4; ++ht) {
#pragma unroll
        for (int v = 0; v < 8; ++v) {
            int t_idx = w * 16 + half * 8 + v;
            int c = h * C_HD + ht * 16 + r;
            attnout[((size_t)n * C_T + t_idx) * (C_H * C_HD) + c] = f2bf(oacc[ht][v]);
        }
    }
}

// ---------------------------------------------------------------------------
// Kernel 4: output projection attnout[n*T][1024] @ Wo^T -> f32 out (B,T,S,D)
// grid (n=512, cb=4 col-blocks of 256)
// ---------------------------------------------------------------------------
__global__ __launch_bounds__(256) void out_kernel(
        const u16* __restrict__ attnout, const u16* __restrict__ wo,
        float* __restrict__ out) {
    const int n  = blockIdx.x;
    const int cb = blockIdx.y;
    const int tid = threadIdx.x, lane = tid & 31, w = tid >> 5;
    const int half = lane >> 4, r = lane & 15;
    const int b = n >> 8, s = n & 255;

    const int LDA = 40, LDW = 40;
    __shared__ u16 At[128 * 40];
    __shared__ u16 Wt[256 * 40];

    v8f acc[16];
#pragma unroll
    for (int nt = 0; nt < 16; ++nt) acc[nt] = (v8f){0,0,0,0,0,0,0,0};

    const int colbase = cb * 256;
    const int arow = tid >> 1, ac0 = (tid & 1) * 16;
    const u16* asrc_base = attnout + (size_t)n * C_T * C_DIM + (size_t)arow * C_DIM + ac0;
    const u16* wsrc_base = wo + (size_t)(colbase + tid) * C_DIM;

    for (int k0 = 0; k0 < C_DIM; k0 += 32) {
        { // A tile 128x32 bf16 (b128 copies)
            const uint4* src = (const uint4*)(asrc_base + k0);
            uint4* dst = (uint4*)&At[arow * LDA + ac0];
            dst[0] = src[0]; dst[1] = src[1];
        }
        { // W tile 256x32 bf16
            const uint4* src = (const uint4*)(wsrc_base + k0);
            uint4* dst = (uint4*)&Wt[tid * LDW];
#pragma unroll
            for (int i = 0; i < 4; ++i) dst[i] = src[i];
        }
        if (k0 + 32 < C_DIM) {
            __builtin_prefetch((const void*)(asrc_base + k0 + 32), 0, 1);
            __builtin_prefetch((const void*)(wsrc_base + k0 + 32), 0, 1);
        }
        __syncthreads();
        v16bf a = ld_frag_a(&At[(w * 16) * LDA], LDA, lane);
#pragma unroll
        for (int nt = 0; nt < 16; ++nt) {
            v16bf bf = ld_frag_b(&Wt[(nt * 16) * LDW], LDW, lane);
            acc[nt] = wmma_bf16(a, bf, acc[nt]);
        }
        __syncthreads();
    }

    // scatter to (B,T,S,D)
#pragma unroll
    for (int nt = 0; nt < 16; ++nt) {
        int col = colbase + nt * 16 + r;
#pragma unroll
        for (int v = 0; v < 8; ++v) {
            int t_idx = w * 16 + half * 8 + v;
            out[(((size_t)b * C_T + t_idx) * C_S + s) * C_DIM + col] = acc[nt][v];
        }
    }
}

// ---------------------------------------------------------------------------
extern "C" void kernel_launch(void* const* d_in, const int* in_sizes, int n_in,
                              void* d_out, int out_size, void* d_ws, size_t ws_size,
                              hipStream_t stream) {
    const float* x  = (const float*)d_in[0];
    const float* Wq = (const float*)d_in[1];
    const float* Wk = (const float*)d_in[2];
    const float* Wv = (const float*)d_in[3];
    const float* Wo = (const float*)d_in[4];
    const float* qg = (const float*)d_in[5];
    const float* kg = (const float*)d_in[6];
    float* out = (float*)d_out;

    char* ws = (char*)d_ws;
    u16* wqkv    = (u16*)(ws + OFF_WQKV);
    u16* wo      = (u16*)(ws + OFF_WO);
    u16* qbuf    = (u16*)(ws + OFF_Q);
    u16* kbuf    = (u16*)(ws + OFF_K);
    u16* vtbuf   = (u16*)(ws + OFF_VT);
    u16* attnout = (u16*)(ws + OFF_AO);

    // 1) weights -> bf16
    {
        size_t tot = 1048576 + 262144 + 262144 + 1048576;
        dim3 grid((unsigned)((tot + 255) / 256));
        prep_kernel<<<grid, 256, 0, stream>>>(Wq, Wk, Wv, Wo, wqkv, wo);
    }
    // 2) QKV + norm + rope
    qkv_kernel<<<dim3(C_N, 6), 256, 0, stream>>>(x, wqkv, qg, kg, qbuf, kbuf, vtbuf);
    // 3) attention
    attn_kernel<<<dim3(C_N, C_H), 256, 0, stream>>>(qbuf, kbuf, vtbuf, attnout);
    // 4) output projection
    out_kernel<<<dim3(C_N, 4), 256, 0, stream>>>(attnout, wo, out);
}
